// CausalMultiHeadSelfAttention_78202764525733
// MI455X (gfx1250) — compile-verified, matrix operand order checked by reading, not run on hardware
//
#include <hip/hip_runtime.h>
#include <hip/hip_bf16.h>

// ---------------------------------------------------------------------------
// Causal multi-head self-attention, MI455X (gfx1250), wave32 + WMMA bf16.
// B=4, S=2048, D=1024, H=16, DK=64.
// Pipeline: f32->bf16 convert -> QKV LDS-tiled NT-GEMM (WMMA + async-to-LDS)
//        -> flash attention (WMMA, async K staging, online softmax)
//        -> output LDS-tiled NT-GEMM (WMMA, f32 out).
// ---------------------------------------------------------------------------

#define B_  4
#define S_  2048
#define D_  1024
#define H_  16
#define DK_ 64
#define M_  (B_ * S_)          // 8192 rows for the projection GEMMs
#define MB_ (M_ / 128)         // 64 block-tiles in M
#define NB_ (D_ / 64)          // 16 block-tiles in N
#define KSTEPS_ (D_ / 32)      // 32 K-steps of 32

typedef __attribute__((ext_vector_type(16))) __bf16 v16bf;
typedef __attribute__((ext_vector_type(8)))  __bf16 v8bf;
typedef __attribute__((ext_vector_type(8)))  float  v8f;
typedef __attribute__((ext_vector_type(4)))  float  v4f;

static __device__ __forceinline__ __bf16 f2bf(float f) {
    // round-to-nearest-even f32 -> bf16
    unsigned u = __float_as_uint(f);
    unsigned r = (u + 0x7FFFu + ((u >> 16) & 1u)) >> 16;
    unsigned short s = (unsigned short)r;
    return __builtin_bit_cast(__bf16, s);
}

// Low 32 bits of a generic pointer into LDS == LDS byte offset (aperture form).
static __device__ __forceinline__ unsigned lds_off(const void* p) {
    return (unsigned)(unsigned long long)p;
}

// CDNA5 async DMA: global -> LDS, 16 bytes per lane, tracked by ASYNCcnt.
static __device__ __forceinline__ void async_ld_b128(void* lds_dst, const void* gsrc) {
    asm volatile("global_load_async_to_lds_b128 %0, %1, off"
                 :: "v"(lds_off(lds_dst)), "v"(gsrc)
                 : "memory");
}

static __device__ __forceinline__ void wait_async0() {
    asm volatile("s_wait_asynccnt 0" ::: "memory");
}

// Load one WMMA 16-bit fragment (A or B side) for this lane.
// ISA layout: lane<16 holds K = {k0..k0+7, k0+16..k0+23} of its row/col,
// lane>=16 holds K = {k0+8..k0+15, k0+24..k0+31}; caller folds the +8 for
// the upper half-wave into `p`, so we always read p[0..7] and p[16..23].
static __device__ __forceinline__ v16bf load_frag(const __bf16* p) {
    v8bf lo = *(const v8bf*)(p);
    v8bf hi = *(const v8bf*)(p + 16);
    v16bf r;
#pragma unroll
    for (int i = 0; i < 8; ++i) { r[i] = lo[i]; r[i + 8] = hi[i]; }
    return r;
}

static __device__ __forceinline__ v8f wmma_bf16(v16bf a, v16bf b, v8f c) {
    return __builtin_amdgcn_wmma_f32_16x16x32_bf16(
               false, a, false, b, (short)0, c, false, false);
}

// ---------------------------------------------------------------------------
// Kernel 0: f32 -> bf16 convert, 4 elements/thread (n divisible by 1024).
// ---------------------------------------------------------------------------
__global__ void cvt_f32_bf16(const float* __restrict__ in,
                             __bf16* __restrict__ out) {
    int i = (blockIdx.x * 256 + threadIdx.x) * 4;
    v4f v = *(const v4f*)(in + i);
    __bf16 o[4];
#pragma unroll
    for (int j = 0; j < 4; ++j) o[j] = f2bf(v[j]);
    *(unsigned long long*)(out + i) = *(unsigned long long*)o;
}

// ---------------------------------------------------------------------------
// Shared core: LDS-tiled NT GEMM block.  Block = 128(M) x 64(N), 8 waves in a
// 4x2 grid, each wave a 32x32 region (2x2 WMMA tiles).  A(128x32) + B(64x32)
// bf16 staged per K-step via global_load_async_to_lds_b128; software-pipelined
// so the async DMA of step k+1 overlaps the WMMAs of step k.
// ---------------------------------------------------------------------------

// stage one K-step of A/B into LDS (3 async b128 per thread)
static __device__ __forceinline__ void stage_ab(__bf16* lA, __bf16* lB,
                                                const __bf16* Ag, const __bf16* Bg,
                                                int m0, int n0, int k0, int tid) {
    int row = tid >> 2;                // 0..63
    int col = (tid & 3) * 8;           // 0,8,16,24
    async_ld_b128(&lA[(row)      * 32 + col], Ag + (size_t)(m0 + row)      * D_ + k0 + col);
    async_ld_b128(&lA[(row + 64) * 32 + col], Ag + (size_t)(m0 + row + 64) * D_ + k0 + col);
    async_ld_b128(&lB[(row)      * 32 + col], Bg + (size_t)(n0 + row)      * D_ + k0 + col);
}

// ---------------------------------------------------------------------------
// Kernel 1: fused QKV projection, y = x @ W^T, Q scaled by 1/sqrt(DK).
// Output layout [B,H,S,DK] (bf16).
// ---------------------------------------------------------------------------
__global__ __launch_bounds__(256)
void qkv_gemm(const __bf16* __restrict__ xb,
              const __bf16* __restrict__ wqb,
              const __bf16* __restrict__ wkb,
              const __bf16* __restrict__ wvb,
              __bf16* __restrict__ Qb,
              __bf16* __restrict__ Kb,
              __bf16* __restrict__ Vb) {
    __shared__ __align__(16) __bf16 lA[128 * 32];   // 8 KB
    __shared__ __align__(16) __bf16 lB[64 * 32];    // 4 KB

    const int tid  = threadIdx.x;
    const int lane = tid & 31;
    const int wave = tid >> 5;
    const int ln   = lane & 15;
    const int half = lane >> 4;
    const int wm   = wave >> 1;        // 0..3
    const int wn   = wave & 1;         // 0..1

    int which = blockIdx.x / (MB_ * NB_);          // 0=Q 1=K 2=V
    int rem   = blockIdx.x % (MB_ * NB_);
    int mB    = rem / NB_;
    int nB    = rem % NB_;
    const int m0 = mB * 128;
    const int n0 = nB * 64;

    const __bf16* W = (which == 0) ? wqb : (which == 1) ? wkb : wvb;

    v8f c[2][2] = {};
    stage_ab(lA, lB, xb, W, m0, n0, 0, tid);

    for (int kt = 0; kt < KSTEPS_; ++kt) {
        wait_async0();
        __syncthreads();
        v16bf a0 = load_frag(&lA[(wm * 32      + ln) * 32 + half * 8]);
        v16bf a1 = load_frag(&lA[(wm * 32 + 16 + ln) * 32 + half * 8]);
        v16bf b0 = load_frag(&lB[(wn * 32      + ln) * 32 + half * 8]);
        v16bf b1 = load_frag(&lB[(wn * 32 + 16 + ln) * 32 + half * 8]);
        __syncthreads();                 // everyone's LDS reads are done
        if (kt + 1 < KSTEPS_)            // overlap next DMA with the WMMAs
            stage_ab(lA, lB, xb, W, m0, n0, (kt + 1) * 32, tid);
        c[0][0] = wmma_bf16(a0, b0, c[0][0]);
        c[0][1] = wmma_bf16(a0, b1, c[0][1]);
        c[1][0] = wmma_bf16(a1, b0, c[1][0]);
        c[1][1] = wmma_bf16(a1, b1, c[1][1]);
    }

    const float scl = (which == 0) ? 0.125f : 1.0f;   // 1/sqrt(64) into Q
    __bf16* Out = (which == 0) ? Qb : (which == 1) ? Kb : Vb;
#pragma unroll
    for (int sm = 0; sm < 2; ++sm) {
#pragma unroll
        for (int sn = 0; sn < 2; ++sn) {
            int n  = n0 + wn * 32 + sn * 16 + ln;
            int h  = n >> 6;
            int dk = n & 63;
#pragma unroll
            for (int r = 0; r < 8; ++r) {
                int m = m0 + wm * 32 + sm * 16 + r + 8 * half;
                int b = m >> 11;                 // m / S_
                int s = m & (S_ - 1);
                Out[(((size_t)(b * H_ + h) * S_) + s) * DK_ + dk] =
                    f2bf(c[sm][sn][r] * scl);
            }
        }
    }
}

// ---------------------------------------------------------------------------
// Kernel 2: causal flash attention.  One block = 8 waves = 128 query rows of
// one (b,h).  K chunks (32x64) staged via async-to-LDS; V staged transposed;
// online softmax with wave32 16-lane shuffles; P bounces through a per-wave
// LDS tile to convert C-layout scores to A-layout for the PV WMMA.
// ---------------------------------------------------------------------------
__global__ __launch_bounds__(256)
void flash_attn(const __bf16* __restrict__ Qb,
                const __bf16* __restrict__ Kb,
                const __bf16* __restrict__ Vb,
                __bf16* __restrict__ attnb) {
    __shared__ __align__(16) __bf16 lK[32 * 64];       // K chunk, row-major
    __shared__ __align__(16) __bf16 lVT[64 * 32];      // V chunk, transposed
    __shared__ __align__(16) __bf16 lP[8][16 * 32];    // per-wave P tile

    const int tid  = threadIdx.x;
    const int lane = tid & 31;
    const int wave = tid >> 5;
    const int ln   = lane & 15;
    const int half = lane >> 4;

    const int bh   = blockIdx.x >> 4;       // (b*H + h)
    const int qblk = blockIdx.x & 15;       // 16 q-blocks of 128 rows
    const int b    = bh / H_;
    const int h    = bh % H_;

    const __bf16* Qh = Qb + (size_t)bh * S_ * DK_;
    const __bf16* Kh = Kb + (size_t)bh * S_ * DK_;
    const __bf16* Vh = Vb + (size_t)bh * S_ * DK_;

    const int q0 = qblk * 128 + wave * 16;  // this wave's 16 query rows

    v16bf qa0 = load_frag(Qh + (size_t)(q0 + ln) * DK_ + 0  + half * 8);
    v16bf qa1 = load_frag(Qh + (size_t)(q0 + ln) * DK_ + 32 + half * 8);

    v8f   acc[4] = {};                      // O tile: 16 x 64 (4 N-tiles)
    float mrow[8], lrow[8];
#pragma unroll
    for (int r = 0; r < 8; ++r) { mrow[r] = -1.0e30f; lrow[r] = 0.0f; }

    const int kvend = qblk * 128 + 128;     // uniform per block (causal bound)

    for (int kv0 = 0; kv0 < kvend; kv0 += 32) {
        __syncthreads();                    // previous chunk fully consumed
        {   // stage K (async DMA) and V (transposed scatter)
            int row = tid >> 3;             // 0..31
            int col = (tid & 7) * 8;        // 0..56
            async_ld_b128(&lK[row * 64 + col],
                          Kh + (size_t)(kv0 + row) * DK_ + col);
            v8bf vv = *(const v8bf*)(Vh + (size_t)(kv0 + row) * DK_ + col);
#pragma unroll
            for (int i = 0; i < 8; ++i) lVT[(col + i) * 32 + row] = vv[i];
        }
        wait_async0();
        __syncthreads();

        if (kv0 <= q0 + 15) {               // wave-uniform: EXEC stays all-1
            // ---- scores: S = Q (16x64) . K^T -> 16x32 (two 16-col frags)
            v8f sc[2] = {{}, {}};
#pragma unroll
            for (int cb = 0; cb < 2; ++cb) {
                v16bf b0 = load_frag(&lK[(cb * 16 + ln) * 64 + 0  + half * 8]);
                v16bf b1 = load_frag(&lK[(cb * 16 + ln) * 64 + 32 + half * 8]);
                sc[cb] = wmma_bf16(qa0, b0, sc[cb]);
                sc[cb] = wmma_bf16(qa1, b1, sc[cb]);
            }

            // ---- causal mask + online softmax (rows live across 16 lanes)
#pragma unroll
            for (int r = 0; r < 8; ++r) {
                int m = r + 8 * half;
                int q = q0 + m;
                float s0 = sc[0][r];
                float s1 = sc[1][r];
                if (kv0 + ln      > q) s0 = -1.0e30f;
                if (kv0 + 16 + ln > q) s1 = -1.0e30f;

                float t = fmaxf(s0, s1);
                t = fmaxf(t, __shfl_xor(t, 1));
                t = fmaxf(t, __shfl_xor(t, 2));
                t = fmaxf(t, __shfl_xor(t, 4));
                t = fmaxf(t, __shfl_xor(t, 8));
                float mn = fmaxf(mrow[r], t);

                float p0 = __expf(s0 - mn);
                float p1 = __expf(s1 - mn);
                float ps = p0 + p1;
                ps += __shfl_xor(ps, 1);
                ps += __shfl_xor(ps, 2);
                ps += __shfl_xor(ps, 4);
                ps += __shfl_xor(ps, 8);

                float corr = __expf(mrow[r] - mn);
                lrow[r] = lrow[r] * corr + ps;
                mrow[r] = mn;
#pragma unroll
                for (int t4 = 0; t4 < 4; ++t4) acc[t4][r] *= corr;

                lP[wave][m * 32 + ln]      = f2bf(p0);
                lP[wave][m * 32 + 16 + ln] = f2bf(p1);
            }

            // ---- O += P (16x32) . V (32x64)  (per-wave LDS ops are in-order)
            v16bf pa = load_frag(&lP[wave][ln * 32 + half * 8]);
#pragma unroll
            for (int t4 = 0; t4 < 4; ++t4) {
                v16bf vb = load_frag(&lVT[(t4 * 16 + ln) * 32 + half * 8]);
                acc[t4] = wmma_bf16(pa, vb, acc[t4]);
            }
        }
    }

    // epilogue: normalize, store bf16 in [B,S,D] (= [B,S,H,DK]) layout
#pragma unroll
    for (int t4 = 0; t4 < 4; ++t4) {
#pragma unroll
        for (int r = 0; r < 8; ++r) {
            int m = r + 8 * half;
            float o = acc[t4][r] / lrow[r];
            attnb[((size_t)b * S_ + q0 + m) * D_ + h * DK_ + t4 * 16 + ln] = f2bf(o);
        }
    }
}

// ---------------------------------------------------------------------------
// Kernel 3: output projection, out = attn @ W_O^T, f32 result.
// ---------------------------------------------------------------------------
__global__ __launch_bounds__(256)
void oproj_gemm(const __bf16* __restrict__ ab,
                const __bf16* __restrict__ wob,
                float* __restrict__ out) {
    __shared__ __align__(16) __bf16 lA[128 * 32];
    __shared__ __align__(16) __bf16 lB[64 * 32];

    const int tid  = threadIdx.x;
    const int lane = tid & 31;
    const int wave = tid >> 5;
    const int ln   = lane & 15;
    const int half = lane >> 4;
    const int wm   = wave >> 1;
    const int wn   = wave & 1;

    int mB = blockIdx.x / NB_;
    int nB = blockIdx.x % NB_;
    const int m0 = mB * 128;
    const int n0 = nB * 64;

    v8f c[2][2] = {};
    stage_ab(lA, lB, ab, wob, m0, n0, 0, tid);

    for (int kt = 0; kt < KSTEPS_; ++kt) {
        wait_async0();
        __syncthreads();
        v16bf a0 = load_frag(&lA[(wm * 32      + ln) * 32 + half * 8]);
        v16bf a1 = load_frag(&lA[(wm * 32 + 16 + ln) * 32 + half * 8]);
        v16bf b0 = load_frag(&lB[(wn * 32      + ln) * 32 + half * 8]);
        v16bf b1 = load_frag(&lB[(wn * 32 + 16 + ln) * 32 + half * 8]);
        __syncthreads();
        if (kt + 1 < KSTEPS_)
            stage_ab(lA, lB, ab, wob, m0, n0, (kt + 1) * 32, tid);
        c[0][0] = wmma_bf16(a0, b0, c[0][0]);
        c[0][1] = wmma_bf16(a0, b1, c[0][1]);
        c[1][0] = wmma_bf16(a1, b0, c[1][0]);
        c[1][1] = wmma_bf16(a1, b1, c[1][1]);
    }

#pragma unroll
    for (int sm = 0; sm < 2; ++sm) {
#pragma unroll
        for (int sn = 0; sn < 2; ++sn) {
#pragma unroll
            for (int r = 0; r < 8; ++r) {
                int m = m0 + wm * 32 + sm * 16 + r + 8 * half;
                int n = n0 + wn * 32 + sn * 16 + ln;
                out[(size_t)m * D_ + n] = c[sm][sn][r];
            }
        }
    }
}

// ---------------------------------------------------------------------------
// Host-side launcher
// ---------------------------------------------------------------------------
extern "C" void kernel_launch(void* const* d_in, const int* in_sizes, int n_in,
                              void* d_out, int out_size, void* d_ws, size_t ws_size,
                              hipStream_t stream) {
    (void)in_sizes; (void)n_in; (void)out_size; (void)ws_size;

    const float* x  = (const float*)d_in[0];
    const float* wq = (const float*)d_in[1];
    const float* wk = (const float*)d_in[2];
    const float* wv = (const float*)d_in[3];
    const float* wo = (const float*)d_in[4];
    float* out = (float*)d_out;

    char* ws = (char*)d_ws;
    const size_t NX = (size_t)B_ * S_ * D_;   // 8388608
    const size_t NW = (size_t)D_ * D_;        // 1048576

    __bf16* xb    = (__bf16*)(ws);                                   // 16 MB
    __bf16* wqb   = (__bf16*)(ws + 16777216);                        //  2 MB
    __bf16* wkb   = (__bf16*)(ws + 18874368);
    __bf16* wvb   = (__bf16*)(ws + 20971520);
    __bf16* wob   = (__bf16*)(ws + 23068672);
    __bf16* Qb    = (__bf16*)(ws + 25165824);                        // 16 MB
    __bf16* Kb    = (__bf16*)(ws + 41943040);                        // 16 MB
    __bf16* Vb    = (__bf16*)(ws + 58720256);                        // 16 MB
    __bf16* attnb = (__bf16*)(ws + 75497472);                        // 16 MB

    cvt_f32_bf16<<<(int)(NX / 1024), 256, 0, stream>>>(x,  xb);
    cvt_f32_bf16<<<(int)(NW / 1024), 256, 0, stream>>>(wq, wqb);
    cvt_f32_bf16<<<(int)(NW / 1024), 256, 0, stream>>>(wk, wkb);
    cvt_f32_bf16<<<(int)(NW / 1024), 256, 0, stream>>>(wv, wvb);
    cvt_f32_bf16<<<(int)(NW / 1024), 256, 0, stream>>>(wv == wo ? wo : wo, wob);

    // QKV: 3 matrices * 64(M) * 16(N) block tiles
    qkv_gemm<<<3 * MB_ * NB_, 256, 0, stream>>>(xb, wqb, wkb, wvb, Qb, Kb, Vb);

    // Attention: B*H * (S/128) blocks
    flash_attn<<<B_ * H_ * (S_ / 128), 256, 0, stream>>>(Qb, Kb, Vb, attnb);

    // Output projection: 64 * 16 block tiles
    oproj_gemm<<<MB_ * NB_, 256, 0, stream>>>(attnb, wob, out);
}
